// Attention_9517647528123
// MI455X (gfx1250) — matrix operand order repaired and checked by reading, not compiled, not run on hardware
//
#include <hip/hip_runtime.h>

#define SQ 2048
#define DMODEL 2048
#define NH 16
#define NKV 4
#define HD 128
#define DD 64
#define NB 64

typedef __attribute__((ext_vector_type(16))) __bf16 v16bf;
typedef __attribute__((ext_vector_type(8)))  float  v8f;

union BF16Frag { unsigned int u[8]; v16bf v; };

#define FINF (__builtin_inff())

__device__ inline unsigned short f2bf(float f) {
  unsigned int u = __float_as_uint(f);
  return (unsigned short)((u + 0x7FFFu + ((u >> 16) & 1u)) >> 16);
}
__device__ inline float bf2f(unsigned short h) {
  return __uint_as_float(((unsigned int)h) << 16);
}
__device__ inline v8f wmma_bf16(v16bf a, v16bf b, v8f c) {
  return __builtin_amdgcn_wmma_f32_16x16x32_bf16(false, a, false, b, (short)0, c, false, false);
}

// gfx1250 async global->LDS (tracked with ASYNCcnt, bypasses VGPRs)
__device__ inline void async_g2l_b128(unsigned int lds_off, const void* gaddr) {
  asm volatile("global_load_async_to_lds_b128 %0, %1, off"
               :: "v"(lds_off), "v"(gaddr) : "memory");
}
__device__ inline void wait_async0() {
  asm volatile("s_wait_asynccnt 0x0" ::: "memory");
}

// ---------------------------------------------------------------- conversions
__global__ void cvt_f32_bf16(const float* __restrict__ src,
                             unsigned short* __restrict__ dst, int n) {
  for (int i = blockIdx.x * blockDim.x + threadIdx.x; i < n;
       i += gridDim.x * blockDim.x)
    dst[i] = f2bf(src[i]);
}

// RoPE + convert: src fp32 [S][nh][128] -> dst bf16, interleaved pairs
__global__ void rope_cvt(const float* __restrict__ src,
                         unsigned short* __restrict__ dst,
                         const float* __restrict__ cosd,
                         const float* __restrict__ sind, int nh) {
  int idx = blockIdx.x * blockDim.x + threadIdx.x;
  int total = SQ * nh * (HD / 2);
  if (idx >= total) return;
  int p = idx % (HD / 2);
  int h = (idx / (HD / 2)) % nh;
  int s = idx / (nh * (HD / 2));
  int base = (s * nh + h) * HD + 2 * p;
  float xr = src[base], xi = src[base + 1];
  float c = cosd[s * (HD / 2) + p], si = sind[s * (HD / 2) + p];
  dst[base]     = f2bf(xr * c - xi * si);
  dst[base + 1] = f2bf(xr * si + xi * c);
}

// V fp32 [S][512] -> Vt bf16 [512][S]
__global__ void cvt_transpose_v(const float* __restrict__ Vf,
                                unsigned short* __restrict__ Vt) {
  int idx = blockIdx.x * blockDim.x + threadIdx.x;
  if (idx >= SQ * NKV * HD) return;
  int s = idx / (NKV * HD);
  int dh = idx % (NKV * HD);
  Vt[dh * SQ + s] = f2bf(Vf[idx]);
}

// --------------------------------------------------------------- bf16 NT GEMM
// C[M][N] = A[M][K] * B[N][K]^T, fp32 out. 128x128 block tile, BK=32, 8 waves.
// Double-buffered LDS fed by async global->LDS DMA loads (ASYNCcnt).
__global__ __launch_bounds__(256) void gemm_bf16_nt(
    const unsigned short* __restrict__ A, const unsigned short* __restrict__ B,
    float* __restrict__ C, int M, int N, int K) {
  __shared__ unsigned short As[2][128 * 32];
  __shared__ unsigned short Bs[2][128 * 32];
  const int tid = threadIdx.x;
  const int w = tid >> 5, lane = tid & 31;
  const int n16 = lane & 15, hi = lane >> 4;
  const int wm = w & 3, wn = w >> 2;
  const int bm = blockIdx.y * 128, bn = blockIdx.x * 128;

  // each thread stages two 16B chunks of A and two of B per K-step
  const int c0 = tid, c1 = tid + 256;
  const int rowA0 = c0 >> 2, chA0 = (c0 & 3) * 8;
  const int rowA1 = c1 >> 2, chA1 = (c1 & 3) * 8;
  const unsigned short* gA0 = A + (size_t)(bm + rowA0) * K + chA0;
  const unsigned short* gA1 = A + (size_t)(bm + rowA1) * K + chA1;
  const unsigned short* gB0 = B + (size_t)(bn + rowA0) * K + chA0;
  const unsigned short* gB1 = B + (size_t)(bn + rowA1) * K + chA1;
  const unsigned int lA0[2] = {
      (unsigned int)(unsigned long long)&As[0][rowA0 * 32 + chA0],
      (unsigned int)(unsigned long long)&As[1][rowA0 * 32 + chA0]};
  const unsigned int lA1[2] = {
      (unsigned int)(unsigned long long)&As[0][rowA1 * 32 + chA1],
      (unsigned int)(unsigned long long)&As[1][rowA1 * 32 + chA1]};
  const unsigned int lB0[2] = {
      (unsigned int)(unsigned long long)&Bs[0][rowA0 * 32 + chA0],
      (unsigned int)(unsigned long long)&Bs[1][rowA0 * 32 + chA0]};
  const unsigned int lB1[2] = {
      (unsigned int)(unsigned long long)&Bs[0][rowA1 * 32 + chA1],
      (unsigned int)(unsigned long long)&Bs[1][rowA1 * 32 + chA1]};

  v8f acc[2][4];
#pragma unroll
  for (int im = 0; im < 2; ++im)
#pragma unroll
    for (int in_ = 0; in_ < 4; ++in_)
      acc[im][in_] = (v8f){0.f, 0.f, 0.f, 0.f, 0.f, 0.f, 0.f, 0.f};

  // prologue: async-stage K-step 0 into buffer 0
  async_g2l_b128(lA0[0], gA0);
  async_g2l_b128(lA1[0], gA1);
  async_g2l_b128(lB0[0], gB0);
  async_g2l_b128(lB1[0], gB1);

  for (int k0 = 0; k0 < K; k0 += 32) {
    const int cur = (k0 >> 5) & 1;
    wait_async0();      // my async DMAs for buffer `cur` have landed in LDS
    __syncthreads();    // everyone's have

    if (k0 + 32 < K) {  // prefetch next K-step into the other buffer
      const int nx = cur ^ 1;
      async_g2l_b128(lA0[nx], gA0 + k0 + 32);
      async_g2l_b128(lA1[nx], gA1 + k0 + 32);
      async_g2l_b128(lB0[nx], gB0 + k0 + 32);
      async_g2l_b128(lB1[nx], gB1 + k0 + 32);
    }

    BF16Frag af[2], bfr[4];
#pragma unroll
    for (int im = 0; im < 2; ++im) {
      int m0 = wm * 32 + im * 16 + n16;
#pragma unroll
      for (int v = 0; v < 8; ++v) {
        int k = hi * 8 + (v < 4 ? 2 * v : 16 + 2 * (v - 4));
        af[im].u[v] = *(const unsigned int*)(&As[cur][m0 * 32 + k]);
      }
    }
#pragma unroll
    for (int in_ = 0; in_ < 4; ++in_) {
      int nn = wn * 64 + in_ * 16 + n16;
#pragma unroll
      for (int v = 0; v < 8; ++v) {
        int k = hi * 8 + (v < 4 ? 2 * v : 16 + 2 * (v - 4));
        bfr[in_].u[v] = *(const unsigned int*)(&Bs[cur][nn * 32 + k]);
      }
    }
#pragma unroll
    for (int im = 0; im < 2; ++im)
#pragma unroll
      for (int in_ = 0; in_ < 4; ++in_)
        acc[im][in_] = wmma_bf16(af[im].v, bfr[in_].v, acc[im][in_]);
    __syncthreads();    // all reads of buffer `cur` done before its next refill
  }

#pragma unroll
  for (int im = 0; im < 2; ++im)
#pragma unroll
    for (int in_ = 0; in_ < 4; ++in_)
#pragma unroll
      for (int r = 0; r < 8; ++r)
        C[(size_t)(bm + wm * 32 + im * 16 + r + hi * 8) * N + bn + wn * 64 +
          in_ * 16 + n16] = acc[im][in_][r];
}

// ------------------------------------------------------- block selection path
// q_d[s][h][D] = sum_d Qb[s][h][d] * Wqd[d][h][D]
__global__ void qd_kernel(const unsigned short* __restrict__ Qb,
                          const float* __restrict__ Wqd,
                          float* __restrict__ qd) {
  int idx = blockIdx.x * blockDim.x + threadIdx.x;
  if (idx >= SQ * NH * DD) return;
  int D = idx % DD;
  int h = (idx / DD) % NH;
  int s = idx / (NH * DD);
  float acc = 0.f;
  for (int d = 0; d < HD; ++d)
    acc += bf2f(Qb[(s * NH + h) * HD + d]) * Wqd[(d * NH + h) * DD + D];
  qd[idx] = acc;
}

// kbm[blk][h][D] = mean_{r<32} sum_d Kb[(32blk+r)][h>>2][d] * Wkd[d][h][D]
__global__ void kbm_kernel(const unsigned short* __restrict__ Kb,
                           const float* __restrict__ Wkd,
                           float* __restrict__ kbm) {
  int idx = blockIdx.x * blockDim.x + threadIdx.x;
  if (idx >= NB * NH * DD) return;
  int D = idx % DD;
  int h = (idx / DD) % NH;
  int blk = idx / (NH * DD);
  int kvh = h >> 2;
  float acc = 0.f;
  for (int r = 0; r < 32; ++r) {
    int s = blk * 32 + r;
    for (int d = 0; d < HD; ++d)
      acc += bf2f(Kb[(s * NKV + kvh) * HD + d]) * Wkd[(d * NH + h) * DD + D];
  }
  kbm[idx] = acc * (1.f / 32.f);
}

// per (h,s): top-3 blocks of qd.kbm + base block -> 64-bit mask
__global__ void topk_kernel(const float* __restrict__ qd,
                            const float* __restrict__ kbm,
                            unsigned long long* __restrict__ mask) {
  int idx = blockIdx.x * blockDim.x + threadIdx.x;
  if (idx >= NH * SQ) return;
  int h = idx / SQ, s = idx % SQ;
  float sc[NB];
  for (int j = 0; j < NB; ++j) {
    float acc = 0.f;
    for (int D = 0; D < DD; ++D)
      acc += qd[(s * NH + h) * DD + D] * kbm[(j * NH + h) * DD + D];
    sc[j] = acc;
  }
  unsigned long long m = 0ull;
  for (int t = 0; t < 3; ++t) {
    float best = -FINF;
    int bi = -1;
    for (int j = 0; j < NB; ++j)
      if (!((m >> j) & 1ull) && sc[j] > best) { best = sc[j]; bi = j; }
    if (bi >= 0) m |= 1ull << bi;
  }
  int bidx = (s + 31) >> 5;
  if (bidx > NB - 1) bidx = NB - 1;
  m |= 1ull << bidx;
  mask[h * SQ + s] = m;
}

// ------------------------------------------------------------ flash attention
// One wave per (head, 16-query tile). Sparse over 32-key blocks.
__global__ __launch_bounds__(256) void flash_attn(
    const unsigned short* __restrict__ Qb,   // [S][16][128]
    const unsigned short* __restrict__ Kb,   // [S][4][128]
    const unsigned short* __restrict__ Vt,   // [4*128][S]
    const unsigned long long* __restrict__ mask,  // [16][S]
    unsigned short* __restrict__ attb) {     // [S][2048]
  __shared__ unsigned short plds[8][16 * 32];
  const int tid = threadIdx.x;
  const int w = tid >> 5;
  const int lane = tid & 31;
  const int n = lane & 15;
  const int hi = lane >> 4;
  const int gw = blockIdx.x * 8 + w;
  const int h = gw >> 7;
  const int q0 = (gw & 127) << 4;
  const int kvh = h >> 2;
  const float SCALE = 0.08838834764831845f;  // 1/sqrt(128)

  // Preload Q fragments (A layout), 4 K-steps of 32 over d=128
  BF16Frag qf[4];
#pragma unroll
  for (int ds_ = 0; ds_ < 4; ++ds_) {
    int base = ((q0 + n) * NH + h) * HD + ds_ * 32;
#pragma unroll
    for (int v = 0; v < 8; ++v) {
      int k = hi * 8 + (v < 4 ? 2 * v : 16 + 2 * (v - 4));
      qf[ds_].u[v] = *(const unsigned int*)(Qb + base + k);
    }
  }

  // Row masks for this lane's 8 rows; wave-uniform union
  unsigned long long rm[8];
  unsigned long long mu = 0ull;
#pragma unroll
  for (int r = 0; r < 8; ++r) {
    rm[r] = mask[h * SQ + q0 + r + hi * 8];
    mu |= rm[r];
  }
  {
    unsigned int lo = (unsigned int)mu, hi32 = (unsigned int)(mu >> 32);
    lo |= __shfl_xor(lo, 16, 32);
    hi32 |= __shfl_xor(hi32, 16, 32);
    mu = ((unsigned long long)hi32 << 32) | lo;
  }

  v8f o[8];
#pragma unroll
  for (int dt = 0; dt < 8; ++dt)
    o[dt] = (v8f){0.f, 0.f, 0.f, 0.f, 0.f, 0.f, 0.f, 0.f};
  float mrow[8], lrow[8];
#pragma unroll
  for (int r = 0; r < 8; ++r) { mrow[r] = -FINF; lrow[r] = 0.f; }

  const int jmax = (q0 + 15) >> 5;
  for (int j = 0; j <= jmax; ++j) {
    if (!((mu >> j) & 1ull)) continue;  // wave-uniform skip

    // scores: two 16x16 tiles over keys j*32 + [0,16) and [16,32)
    v8f s0 = (v8f){0.f, 0.f, 0.f, 0.f, 0.f, 0.f, 0.f, 0.f};
    v8f s1 = s0;
#pragma unroll
    for (int ds_ = 0; ds_ < 4; ++ds_) {
      BF16Frag b0, b1;
      int key0 = j * 32 + n;
      int base0 = (key0 * NKV + kvh) * HD + ds_ * 32;
      int base1 = ((key0 + 16) * NKV + kvh) * HD + ds_ * 32;
#pragma unroll
      for (int v = 0; v < 8; ++v) {
        int k = hi * 8 + (v < 4 ? 2 * v : 16 + 2 * (v - 4));
        b0.u[v] = *(const unsigned int*)(Kb + base0 + k);
        b1.u[v] = *(const unsigned int*)(Kb + base1 + k);
      }
      s0 = wmma_bf16(qf[ds_].v, b0.v, s0);
      s1 = wmma_bf16(qf[ds_].v, b1.v, s1);
    }

    // branch-free online softmax (per row, 16-lane shuffles)
#pragma unroll
    for (int r = 0; r < 8; ++r) {
      int row = q0 + r + hi * 8;
      bool bit = (rm[r] >> j) & 1ull;
      int key0 = j * 32 + n, key1 = key0 + 16;
      float a0 = (bit && key0 <= row) ? s0[r] * SCALE : -FINF;
      float a1 = (bit && key1 <= row) ? s1[r] * SCALE : -FINF;
      float vm = fmaxf(a0, a1);
      vm = fmaxf(vm, __shfl_xor(vm, 1, 32));
      vm = fmaxf(vm, __shfl_xor(vm, 2, 32));
      vm = fmaxf(vm, __shfl_xor(vm, 4, 32));
      vm = fmaxf(vm, __shfl_xor(vm, 8, 32));
      float mnew = fmaxf(mrow[r], vm);
      float me = fmaxf(mnew, -1e30f);          // keeps exp args finite->0
      float alpha = __expf(mrow[r] - me);
      float p0 = __expf(a0 - me);
      float p1 = __expf(a1 - me);
      float ps = p0 + p1;
      ps += __shfl_xor(ps, 1, 32);
      ps += __shfl_xor(ps, 2, 32);
      ps += __shfl_xor(ps, 4, 32);
      ps += __shfl_xor(ps, 8, 32);
      lrow[r] = lrow[r] * alpha + ps;
      mrow[r] = mnew;
#pragma unroll
      for (int dt = 0; dt < 8; ++dt) o[dt][r] = o[dt][r] * alpha;
      plds[w][(r + hi * 8) * 32 + n] = f2bf(p0);
      plds[w][(r + hi * 8) * 32 + 16 + n] = f2bf(p1);
    }

    // reload P in A layout (C->A relayout via LDS)
    BF16Frag pf;
#pragma unroll
    for (int v = 0; v < 8; ++v) {
      int k = hi * 8 + (v < 4 ? 2 * v : 16 + 2 * (v - 4));
      pf.u[v] = *(const unsigned int*)(&plds[w][n * 32 + k]);
    }
    // PV: O[16x128] += P[16x32] * V[32x128]
#pragma unroll
    for (int dt = 0; dt < 8; ++dt) {
      BF16Frag vf;
      const unsigned short* vp = Vt + (kvh * HD + dt * 16 + n) * SQ + j * 32;
#pragma unroll
      for (int v = 0; v < 8; ++v)
        vf.u[v] = *(const unsigned int*)(vp + hi * 8 + (v < 4 ? 2 * v : 16 + 2 * (v - 4)));
      o[dt] = wmma_bf16(pf.v, vf.v, o[dt]);
    }
  }

#pragma unroll
  for (int r = 0; r < 8; ++r) {
    float inv = lrow[r] > 0.f ? 1.f / lrow[r] : 0.f;
    int row = q0 + r + hi * 8;
#pragma unroll
    for (int dt = 0; dt < 8; ++dt)
      attb[row * DMODEL + h * HD + dt * 16 + n] = f2bf(o[dt][r] * inv);
  }
}

// -------------------------------------------------------------------- launch
extern "C" void kernel_launch(void* const* d_in, const int* in_sizes, int n_in,
                              void* d_out, int out_size, void* d_ws,
                              size_t ws_size, hipStream_t stream) {
  const float* x   = (const float*)d_in[0];
  const float* wq  = (const float*)d_in[1];
  const float* wk  = (const float*)d_in[2];
  const float* wv  = (const float*)d_in[3];
  const float* wo  = (const float*)d_in[4];
  const float* Wqd = (const float*)d_in[5];
  const float* Wkd = (const float*)d_in[6];
  const float* fc  = (const float*)d_in[7];
  const float* fs  = (const float*)d_in[8];
  float* out = (float*)d_out;

  char* base = (char*)d_ws;
  size_t off = 0;
  auto carve = [&](size_t bytes) {
    char* p = base + off;
    off = (off + bytes + 255) & ~(size_t)255;
    return p;
  };
  unsigned short* xb  = (unsigned short*)carve((size_t)SQ * DMODEL * 2);
  unsigned short* wqb = (unsigned short*)carve((size_t)DMODEL * DMODEL * 2);
  unsigned short* wkb = (unsigned short*)carve((size_t)NKV * HD * DMODEL * 2);
  unsigned short* wvb = (unsigned short*)carve((size_t)NKV * HD * DMODEL * 2);
  unsigned short* wob = (unsigned short*)carve((size_t)DMODEL * DMODEL * 2);
  float* Qf = (float*)carve((size_t)SQ * DMODEL * 4);
  float* Kf = (float*)carve((size_t)SQ * NKV * HD * 4);
  float* Vf = (float*)carve((size_t)SQ * NKV * HD * 4);
  unsigned short* Qb = (unsigned short*)carve((size_t)SQ * DMODEL * 2);
  unsigned short* Kb = (unsigned short*)carve((size_t)SQ * NKV * HD * 2);
  unsigned short* Vt = (unsigned short*)carve((size_t)SQ * NKV * HD * 2);
  unsigned short* attb = (unsigned short*)carve((size_t)SQ * DMODEL * 2);
  float* qd  = (float*)carve((size_t)SQ * NH * DD * 4);
  float* kbm = (float*)carve((size_t)NB * NH * DD * 4);
  unsigned long long* msk =
      (unsigned long long*)carve((size_t)NH * SQ * 8);

  const int T = 256;
  cvt_f32_bf16<<<(SQ * DMODEL + T - 1) / T, T, 0, stream>>>(x, xb, SQ * DMODEL);
  cvt_f32_bf16<<<(DMODEL * DMODEL + T - 1) / T, T, 0, stream>>>(wq, wqb, DMODEL * DMODEL);
  cvt_f32_bf16<<<(NKV * HD * DMODEL + T - 1) / T, T, 0, stream>>>(wk, wkb, NKV * HD * DMODEL);
  cvt_f32_bf16<<<(NKV * HD * DMODEL + T - 1) / T, T, 0, stream>>>(wv, wvb, NKV * HD * DMODEL);
  cvt_f32_bf16<<<(DMODEL * DMODEL + T - 1) / T, T, 0, stream>>>(wo, wob, DMODEL * DMODEL);

  gemm_bf16_nt<<<dim3(DMODEL / 128, SQ / 128), T, 0, stream>>>(
      xb, wqb, Qf, SQ, DMODEL, DMODEL);
  gemm_bf16_nt<<<dim3(NKV * HD / 128, SQ / 128), T, 0, stream>>>(
      xb, wkb, Kf, SQ, NKV * HD, DMODEL);
  gemm_bf16_nt<<<dim3(NKV * HD / 128, SQ / 128), T, 0, stream>>>(
      xb, wvb, Vf, SQ, NKV * HD, DMODEL);

  rope_cvt<<<(SQ * NH * HD / 2 + T - 1) / T, T, 0, stream>>>(Qf, Qb, fc, fs, NH);
  rope_cvt<<<(SQ * NKV * HD / 2 + T - 1) / T, T, 0, stream>>>(Kf, Kb, fc, fs, NKV);
  cvt_transpose_v<<<(SQ * NKV * HD + T - 1) / T, T, 0, stream>>>(Vf, Vt);

  qd_kernel<<<(SQ * NH * DD + T - 1) / T, T, 0, stream>>>(Qb, Wqd, qd);
  kbm_kernel<<<(NB * NH * DD + T - 1) / T, T, 0, stream>>>(Kb, Wkd, kbm);
  topk_kernel<<<(NH * SQ + T - 1) / T, T, 0, stream>>>(qd, kbm, msk);

  flash_attn<<<(NH * (SQ / 16)) / 8, T, 0, stream>>>(Qb, Kb, Vt, msk, attb);

  gemm_bf16_nt<<<dim3(DMODEL / 128, SQ / 128), T, 0, stream>>>(
      attb, wob, out, SQ, DMODEL, DMODEL);
}
